// Modnet2_13572096655547
// MI455X (gfx1250) — matrix operand order repaired and checked by reading, not compiled, not run on hardware
//
#include <hip/hip_runtime.h>

// ---------------------------------------------------------------------------
// Problem constants (from reference setup_inputs): N=2048, M=512, D=2, H=128.
// out[i] = sum_j y[j] * MLP(concat(input[i], quad_x[j]))
//   layer1 separable: A1[i,h] = input[i,:] @ W1[0:2, h]
//                     B1[j,h] = quad_x[j,:] @ W1[2:4, h] + b1[h]
//   h1 = tanh(A1[i]+B1[j]);  h2 = tanh(h1 @ W2 + b2);  g = h2 @ W3 + b3
// Dominant GEMM (h1 @ W2) runs on v_wmma_f32_16x16x32_f16.
// ---------------------------------------------------------------------------

#define N_IN   2048
#define M_Q    512
#define H_DIM  128
#define PI_F   3.14159265358979323846f

typedef __attribute__((ext_vector_type(16))) _Float16 v16h;
typedef __attribute__((ext_vector_type(8)))  float    v8f;
typedef __attribute__((ext_vector_type(4)))  float    v4f;

// gfx1250 hardware tanh (TRANS pipe, co-executes with WMMA/XDL).
__device__ __forceinline__ float fast_tanh(float x) {
  float r;
  asm("v_tanh_f32 %0, %1" : "=v"(r) : "v"(x));
  return r;
}

// Reorder map for the f16 A-matrix 16x32 fragment (ISA 7.12.2):
//   vector elem p, lane-half g (lane>=16):  k = 32c + (p&7) + ((p>>3)<<4) + (g<<3)
// We store A1/B1 rows so that [32c + 16g + p] holds value at original k.
// => main kernel reads 16 consecutive floats per (chunk, lane-half).

// ---------------------------------------------------------------------------
// Kernel 1: precompute A1r (reordered), B1r (reordered), y, and W2 as f16
// fragments in WMMA B-matrix order: W2h[((t*4+c)*32 + lane)*16 + p]
//   n = 16t + (lane&15),  k = 32c + (p&7) + ((p>>3)<<4) + 8*(lane>>4)   (B 32x16 layout)
// ---------------------------------------------------------------------------
__global__ void precompute_kernel(const float* __restrict__ input,
                                  const float* __restrict__ quad_x,
                                  const float* __restrict__ eq_param,
                                  const float* __restrict__ W1,
                                  const float* __restrict__ b1,
                                  const float* __restrict__ W2,
                                  float* __restrict__ A1r,
                                  float* __restrict__ B1r,
                                  float* __restrict__ yv,
                                  _Float16* __restrict__ W2h) {
  int tid = blockIdx.x * blockDim.x + threadIdx.x;

  if (tid < N_IN * H_DIM) {                       // A1r
    int i = tid >> 7, pos = tid & 127;
    int c = pos >> 5, r = pos & 31, g = r >> 4, p = r & 15;
    int k = 32 * c + (p & 7) + ((p >> 3) << 4) + (g << 3);
    A1r[tid] = input[i * 2 + 0] * W1[0 * H_DIM + k]
             + input[i * 2 + 1] * W1[1 * H_DIM + k];
    return;
  }
  tid -= N_IN * H_DIM;

  if (tid < M_Q * H_DIM) {                        // B1r
    int j = tid >> 7, pos = tid & 127;
    int c = pos >> 5, r = pos & 31, g = r >> 4, p = r & 15;
    int k = 32 * c + (p & 7) + ((p >> 3) << 4) + (g << 3);
    B1r[tid] = quad_x[j * 2 + 0] * W1[2 * H_DIM + k]
             + quad_x[j * 2 + 1] * W1[3 * H_DIM + k] + b1[k];
    return;
  }
  tid -= M_Q * H_DIM;

  if (tid < M_Q) {                                // y[j] = prod sin(pi*e*qx)
    float e = eq_param[0];
    yv[tid] = sinf(PI_F * e * quad_x[tid * 2 + 0]) *
              sinf(PI_F * e * quad_x[tid * 2 + 1]);
    return;
  }
  tid -= M_Q;

  if (tid < 8 * 4 * 32 * 16) {                    // W2 -> f16 B-fragments
    int p = tid & 15;
    int l = (tid >> 4) & 31;
    int c = (tid >> 9) & 3;
    int t = tid >> 11;
    int n = 16 * t + (l & 15);
    int k = 32 * c + (p & 7) + ((p >> 3) << 4) + 8 * (l >> 4);
    W2h[tid] = (_Float16)W2[k * H_DIM + n];
  }
}

// ---------------------------------------------------------------------------
// Kernel 2: one wave (32 threads) per output row i.
// W2 fragments + A1 row stay in registers; loop over 32 blocks of 16 j-rows.
// Per block: 4 A-fragments (tanh fused), 32 x v_wmma_f32_16x16x32_f16,
// then layer-3 + quadrature fused on the C accumulator layout.
// ---------------------------------------------------------------------------
__global__ void __launch_bounds__(32) modnet_main(
    const float* __restrict__ A1r, const float* __restrict__ B1r,
    const float* __restrict__ yv,  const v16h*  __restrict__ W2h,
    const float* __restrict__ b2,  const float* __restrict__ W3,
    const float* __restrict__ b3p, float* __restrict__ out) {
  const int i    = blockIdx.x;        // output row
  const int lane = threadIdx.x;       // 0..31
  const int gg   = lane >> 4;         // lane half (K-group / row-group select)
  const int nn   = lane & 15;         // column within tile

  // A1 row for this i, already fragment-ordered: a1[c*16+p]
  float a1[64];
#pragma unroll
  for (int c = 0; c < 4; ++c)
#pragma unroll
    for (int q = 0; q < 4; ++q) {
      v4f t4 = *(const v4f*)(A1r + i * H_DIM + 32 * c + 16 * gg + 4 * q);
#pragma unroll
      for (int e = 0; e < 4; ++e) a1[c * 16 + 4 * q + e] = t4[e];
    }

  // W2 B-fragments: [column-tile t][K-chunk c], one 32B load per lane each
  v16h Bf[8][4];
#pragma unroll
  for (int t = 0; t < 8; ++t)
#pragma unroll
    for (int c = 0; c < 4; ++c)
      Bf[t][c] = W2h[(t * 4 + c) * 32 + lane];

  // Per-lane layer-2 bias and layer-3 weight for each column tile
  float b2c[8], w3c[8];
#pragma unroll
  for (int t = 0; t < 8; ++t) {
    b2c[t] = b2[16 * t + nn];
    w3c[t] = W3[16 * t + nn];
  }

  float acc = 0.0f, ysum = 0.0f;
  const v8f vzero = {};

#pragma unroll 1
  for (int jb = 0; jb < M_Q; jb += 16) {
    v8f C[8];
#pragma unroll
    for (int t = 0; t < 8; ++t) C[t] = vzero;

    // Build 4 A-fragments: h1 = tanh(A1[i] + B1[j]) in f16, fragment order
    v16h A[4];
    const float* brow = B1r + (jb + nn) * H_DIM + 16 * gg;
#pragma unroll
    for (int c = 0; c < 4; ++c) {
      v16h af;
#pragma unroll
      for (int q = 0; q < 4; ++q) {
        v4f b4 = *(const v4f*)(brow + 32 * c + 4 * q);
#pragma unroll
        for (int e = 0; e < 4; ++e)
          af[4 * q + e] = (_Float16)fast_tanh(a1[c * 16 + 4 * q + e] + b4[e]);
      }
      A[c] = af;
    }

    // 8 independent accumulation chains x 4 K-steps
#pragma unroll
    for (int c = 0; c < 4; ++c)
#pragma unroll
      for (int t = 0; t < 8; ++t)
        C[t] = __builtin_amdgcn_wmma_f32_16x16x32_f16(
            false, A[c], false, Bf[t][c], (short)0, C[t], false, false);

    // Quadrature weights for the 8 rows this lane-half owns
    float yl[8];
#pragma unroll
    for (int q = 0; q < 2; ++q) {
      v4f t4 = *(const v4f*)(yv + jb + 8 * gg + 4 * q);
#pragma unroll
      for (int e = 0; e < 4; ++e) yl[4 * q + e] = t4[e];
    }

    // Fused layer-3 + quadrature sum on C layout:
    // C[t][v] = h2pre(row = v + 8*gg, col = 16t + nn)
#pragma unroll
    for (int t = 0; t < 8; ++t)
#pragma unroll
      for (int v = 0; v < 8; ++v)
        acc += yl[v] * fast_tanh(C[t][v] + b2c[t]) * w3c[t];

    if (nn == 0) {  // lanes 0 & 16 together cover all 16 rows exactly once
#pragma unroll
      for (int v = 0; v < 8; ++v) ysum += yl[v];
    }
  }

  // Cross-lane reduction (wave32)
#pragma unroll
  for (int off = 16; off >= 1; off >>= 1) {
    acc  += __shfl_xor(acc,  off, 32);
    ysum += __shfl_xor(ysum, off, 32);
  }
  if (lane == 0) out[i] = acc + b3p[0] * ysum;
}

// ---------------------------------------------------------------------------
// Launcher. Workspace layout (256B-aligned sections):
//   A1r: 2048*128 f32 = 1,048,576 B   @ 0
//   B1r:  512*128 f32 =   262,144 B   @ 1,048,576
//   y  :  512 f32     =     2,048 B   @ 1,310,720
//   W2h: 16384 f16    =    32,768 B   @ 1,312,768      (total ~1.29 MB)
// ---------------------------------------------------------------------------
extern "C" void kernel_launch(void* const* d_in, const int* in_sizes, int n_in,
                              void* d_out, int out_size, void* d_ws, size_t ws_size,
                              hipStream_t stream) {
  (void)in_sizes; (void)n_in; (void)out_size; (void)ws_size;
  const float* input  = (const float*)d_in[0];
  const float* quad_x = (const float*)d_in[1];
  const float* eq     = (const float*)d_in[2];
  const float* W1     = (const float*)d_in[3];
  const float* b1     = (const float*)d_in[4];
  const float* W2     = (const float*)d_in[5];
  const float* b2     = (const float*)d_in[6];
  const float* W3     = (const float*)d_in[7];
  const float* b3     = (const float*)d_in[8];
  float* out = (float*)d_out;

  char* ws = (char*)d_ws;
  float*    A1r = (float*)(ws + 0);
  float*    B1r = (float*)(ws + 1048576);
  float*    yv  = (float*)(ws + 1310720);
  _Float16* W2h = (_Float16*)(ws + 1312768);

  const int pre_total = N_IN * H_DIM + M_Q * H_DIM + M_Q + 8 * 4 * 32 * 16;
  precompute_kernel<<<(pre_total + 255) / 256, 256, 0, stream>>>(
      input, quad_x, eq, W1, b1, W2, A1r, B1r, yv, W2h);

  modnet_main<<<N_IN, 32, 0, stream>>>(A1r, B1r, yv, (const v16h*)W2h,
                                       b2, W3, b3, out);
}
